// UncertaintyAwareAttention_37580963840048
// MI455X (gfx1250) — matrix operand order repaired and checked by reading, not compiled, not run on hardware
//
#include <hip/hip_runtime.h>
#include <math.h>

// ---------------------------------------------------------------------------
// UncertaintyAwareAttention for MI455X (gfx1250, wave32, WMMA).
//
// Shapes: B=4, S=2048, E=1024, H=16, HD=64.  d_out = output(B,S,E) fp32
// followed by attn(B,H,S,S) fp32 (1 GB -> the dominant memory cost, ~46us
// at 23.3 TB/s HBM).  Everything else is engineered to stay under that:
//  * all GEMMs in f16 WMMA (v_wmma_f32_16x16x32_f16), f32 accumulate
//  * f16 Q/K/V (48 MB) is L2-resident (192 MB L2) -> WMMA fragments are
//    loaded straight from global and hit L2; scores are *recomputed* in
//    3 passes (max / sum / normalize+PV) instead of spilling raw scores,
//    saving 2 GB of HBM traffic vs a store-and-reload softmax.
//  * the uncertainty bias is a per-(b,h) scalar added to every score:
//    softmax is shift-invariant, so it changes neither attn nor output
//    and is (exactly) skipped.
// ---------------------------------------------------------------------------

typedef _Float16 h16;
typedef __attribute__((ext_vector_type(8)))  _Float16 v8h;
typedef __attribute__((ext_vector_type(16))) _Float16 v16h;
typedef __attribute__((ext_vector_type(8)))  float    v8f;

#define B_  4
#define S_  2048
#define E_  1024
#define H_  16
#define HD_ 64
#define M_  (B_ * S_)   // 8192 tokens total

// ---- WMMA fragment loaders (contiguous k order, same in A and B) ----------------------
// A operand, 16x32 f16 (M x K), source row-major with row stride ld:
//   lane r=lane&15 holds row r; elements = K h*16 .. h*16+15 (contiguous)
//   (h = lane>>4)
static __device__ __forceinline__ v16h load_fragA(const h16* p, int ld) {
  const int lane = threadIdx.x & 31;
  const int r = lane & 15, hh = lane >> 4;
  return *(const v16h*)(p + (size_t)r * ld + hh * 16);
}
// B operand, 32x16 f16 (K x N), source is row-major (N, K) with stride ld:
//   lane r holds column N=r; elements = K hh*16 .. hh*16+15 (contiguous)
static __device__ __forceinline__ v16h load_fragB(const h16* p, int ld) {
  const int lane = threadIdx.x & 31;
  const int r = lane & 15, hh = lane >> 4;
  return *(const v16h*)(p + (size_t)r * ld + hh * 16);
}
// C/D layout: VGPR v holds row (v + 8*(lane>>4)), col (lane&15).

static __device__ __forceinline__ v8f wmma_f16(v16h a, v16h b, v8f c) {
  return __builtin_amdgcn_wmma_f32_16x16x32_f16(false, a, false, b,
                                                (short)0, c, false, false);
}

// ---- fp32 -> f16 elementwise convert ---------------------------------------
__global__ void uaa_conv_f16_kernel(const float* __restrict__ in,
                                    h16* __restrict__ out, int n) {
  int i = blockIdx.x * blockDim.x + threadIdx.x;
  if (i < n) out[i] = (h16)in[i];
}

// ---- GEMM: C = A(8192x1024) @ W(1024x1024)^T + bias ------------------------
// mode 0: out f16, head-major   (B,H,S,HD)   (Q, K projections)
// mode 1: out f16, head-transposed (B,H,HD,S) (V projection -> PV B operand)
// mode 2: out f32, plain (M,E)                (output projection)
// 256 threads = 8 waves; block tile 128(M) x 64(N); wave tile 32x32.
__global__ __launch_bounds__(256)
void uaa_gemm_kernel(const h16* __restrict__ A, const h16* __restrict__ W,
                     const float* __restrict__ bias, void* outp, int mode) {
  const int lane = threadIdx.x & 31;
  const int wid  = threadIdx.x >> 5;
  const int wm = wid & 3, wn = wid >> 2;
  const int rm = blockIdx.y * 128 + wm * 32;
  const int cn = blockIdx.x * 64  + wn * 32;

  v8f cc[2][2] = {};
  for (int k0 = 0; k0 < E_; k0 += 32) {
    v16h a0 = load_fragA(A + (size_t)rm * E_ + k0, E_);
    v16h a1 = load_fragA(A + (size_t)(rm + 16) * E_ + k0, E_);
    v16h b0 = load_fragB(W + (size_t)cn * E_ + k0, E_);
    v16h b1 = load_fragB(W + (size_t)(cn + 16) * E_ + k0, E_);
    cc[0][0] = wmma_f16(a0, b0, cc[0][0]);
    cc[0][1] = wmma_f16(a0, b1, cc[0][1]);
    cc[1][0] = wmma_f16(a1, b0, cc[1][0]);
    cc[1][1] = wmma_f16(a1, b1, cc[1][1]);
  }

  const int r = lane & 15, hh = lane >> 4;
#pragma unroll
  for (int i = 0; i < 2; ++i)
#pragma unroll
    for (int j = 0; j < 2; ++j)
#pragma unroll
      for (int v = 0; v < 8; ++v) {
        const int m = rm + i * 16 + v + 8 * hh;     // token row
        const int n = cn + j * 16 + r;              // output channel
        const float val = cc[i][j][v] + bias[n];
        const int b = m >> 11, s = m & (S_ - 1);    // S = 2048
        const int hd = n >> 6, d = n & (HD_ - 1);
        if (mode == 0) {
          ((h16*)outp)[(((size_t)b * H_ + hd) * S_ + s) * HD_ + d] = (h16)val;
        } else if (mode == 1) {
          ((h16*)outp)[(((size_t)b * H_ + hd) * HD_ + d) * S_ + s] = (h16)val;
        } else {
          ((float*)outp)[(size_t)m * E_ + n] = val;
        }
      }
}

// ---- fused flash-style attention (3 recompute passes) ----------------------
// grid (S/64, B*H), 128 threads = 4 waves, one wave = 16 query rows.
// pass1: per-lane column max  -> half-wave shuffle reduce to row max
// pass2: per-lane exp-sum     -> reduce to row sum
// pass3: p = exp(s-max)/sum; write attn (1 GB); stage p tile in LDS to
//        re-layout C->A fragments; WMMA P @ V^T -> attended (f16).
__global__ __launch_bounds__(128)
void uaa_attn_kernel(const h16* __restrict__ qh, const h16* __restrict__ kh,
                     const h16* __restrict__ vT,
                     const unsigned char* __restrict__ mask,
                     h16* __restrict__ attended, float* __restrict__ attn) {
  alignas(32) __shared__ h16 smP[4 * 16 * 32];   // per-wave 16x32 P tile

  const int lane = threadIdx.x & 31;
  const int wid  = threadIdx.x >> 5;
  const int bh   = blockIdx.y;
  const int b    = bh >> 4;           // H = 16
  const int head = bh & 15;
  const int q0   = blockIdx.x * 64 + wid * 16;   // query row base (within S)
  const int r = lane & 15, hh = lane >> 4;
  const float scale = 0.125f;         // 1/sqrt(64)

  const h16* qbase = qh + ((size_t)bh * S_ + q0) * HD_;
  const h16* kbase = kh + (size_t)bh * S_ * HD_;
  const h16* vbase = vT + (size_t)bh * HD_ * S_;
  const unsigned char* mrow = mask + (size_t)b * S_;

  const v16h aq0 = load_fragA(qbase, HD_);
  const v16h aq1 = load_fragA(qbase + 32, HD_);

  auto score = [&](int n0) -> v8f {
    v8f s = {};
    s = wmma_f16(aq0, load_fragB(kbase + (size_t)n0 * HD_, HD_), s);
    s = wmma_f16(aq1, load_fragB(kbase + (size_t)n0 * HD_ + 32, HD_), s);
    return s;
  };

  float mx[8], sm[8];
#pragma unroll
  for (int v = 0; v < 8; ++v) { mx[v] = -INFINITY; sm[v] = 0.0f; }

  // ---- pass 1: row max -----------------------------------------------------
  for (int nt = 0; nt < S_ / 16; ++nt) {
    const int n0 = nt * 16;
    v8f s = score(n0);
    const bool mk = mrow[n0 + r] != 0;
#pragma unroll
    for (int v = 0; v < 8; ++v) {
      float x = mk ? -INFINITY : s[v] * scale;
      mx[v] = fmaxf(mx[v], x);
    }
  }
#pragma unroll
  for (int v = 0; v < 8; ++v)
#pragma unroll
    for (int off = 1; off < 16; off <<= 1)
      mx[v] = fmaxf(mx[v], __shfl_xor(mx[v], off, 16));

  // ---- pass 2: row sum of exp(s - max) ------------------------------------
  for (int nt = 0; nt < S_ / 16; ++nt) {
    const int n0 = nt * 16;
    v8f s = score(n0);
    const bool mk = mrow[n0 + r] != 0;
#pragma unroll
    for (int v = 0; v < 8; ++v) {
      float x = mk ? -INFINITY : s[v] * scale;
      sm[v] += expf(x - mx[v]);
    }
  }
  float rinv[8];
#pragma unroll
  for (int v = 0; v < 8; ++v) {
#pragma unroll
    for (int off = 1; off < 16; off <<= 1)
      sm[v] += __shfl_xor(sm[v], off, 16);
    rinv[v] = 1.0f / sm[v];
  }

  // ---- pass 3: attn write + P @ V^T ---------------------------------------
  v8f acc[4] = {};
  h16* lme = &smP[wid * 512];
  for (int nt2 = 0; nt2 < S_ / 32; ++nt2) {
#pragma unroll
    for (int sub = 0; sub < 2; ++sub) {
      const int n0 = nt2 * 32 + sub * 16;
      v8f s = score(n0);
      const bool mk = mrow[n0 + r] != 0;
#pragma unroll
      for (int v = 0; v < 8; ++v) {
        float x = mk ? -INFINITY : s[v] * scale;
        float p = expf(x - mx[v]) * rinv[v];
        const int row = q0 + v + 8 * hh;
        attn[((size_t)bh * S_ + row) * S_ + (n0 + r)] = p;
        lme[(v + 8 * hh) * 32 + sub * 16 + r] = (h16)p;   // C->A relayout
      }
    }
    // same-wave LDS is in-order (DScnt) -> load sees this wave's stores
    v16h ap = load_fragA(lme, 32);
#pragma unroll
    for (int dt = 0; dt < 4; ++dt)
      acc[dt] = wmma_f16(ap,
                         load_fragB(vbase + (size_t)(dt * 16) * S_ + nt2 * 32, S_),
                         acc[dt]);
  }

#pragma unroll
  for (int dt = 0; dt < 4; ++dt)
#pragma unroll
    for (int v = 0; v < 8; ++v) {
      const int srow = q0 + v + 8 * hh;
      const size_t g = (size_t)b * S_ + srow;
      attended[g * E_ + head * HD_ + dt * 16 + r] = (h16)acc[dt][v];
    }
}

// ---- gated residual: out = sigmoid(pre.Wg + bg) * pre + (1-g) * query ------
__global__ __launch_bounds__(256)
void uaa_gate_kernel(const float* __restrict__ pre, const float* __restrict__ Wg,
                     const float* __restrict__ bg, const float* __restrict__ query,
                     float* __restrict__ out) {
  __shared__ float red[256];
  const int row = blockIdx.x;
  const int tid = threadIdx.x;
  const float* pr = pre + (size_t)row * E_;
  const float* qr = query + (size_t)row * E_;
  float part = 0.0f;
  for (int e = tid; e < E_; e += 256) part += pr[e] * Wg[e];
  red[tid] = part;
  __syncthreads();
  for (int st = 128; st > 0; st >>= 1) {
    if (tid < st) red[tid] += red[tid + st];
    __syncthreads();
  }
  const float g = 1.0f / (1.0f + expf(-(red[0] + bg[0])));
  for (int e = tid; e < E_; e += 256)
    out[(size_t)row * E_ + e] = g * pr[e] + (1.0f - g) * qr[e];
}

// ---------------------------------------------------------------------------
extern "C" void kernel_launch(void* const* d_in, const int* in_sizes, int n_in,
                              void* d_out, int out_size, void* d_ws, size_t ws_size,
                              hipStream_t stream) {
  (void)in_sizes; (void)n_in; (void)out_size; (void)ws_size;

  const float* query = (const float*)d_in[0];
  const unsigned char* mask = (const unsigned char*)d_in[4];
  // d_in: 5 Wq, 6 bq, 7 Wk, 8 bk, 9 Wv, 10 bv, 15 Wo, 16 bo, 17 Wg, 18 bg
  // (3 uncertainty, 11..14 Wuq/buq/Wuk/buk are exactly irrelevant: softmax
  //  shift-invariance removes the per-(b,h) scalar score bias.)

  char* wsb = (char*)d_ws;
  const size_t MB = 1ull << 20;
  h16* qh  = (h16*)(wsb);             // 16 MB  (B,H,S,HD)   dead after attn
  h16* kh  = (h16*)(wsb + 16 * MB);   // 16 MB  (B,H,S,HD)   dead after attn
  h16* vT  = (h16*)(wsb + 32 * MB);   // 16 MB  (B,H,HD,S)   dead after attn
  h16* x16 = (h16*)(wsb + 48 * MB);   // 16 MB  f16 staging, then attended
  h16* w16 = (h16*)(wsb + 64 * MB);   // 2 MB   f16 weight staging
  float* pre = (float*)(wsb);         // 32 MB  aliases qh+kh (dead by then)

  float* out  = (float*)d_out;
  float* attn = out + (size_t)M_ * E_;   // 268,435,456 floats

  const int nW = E_ * E_, nX = M_ * E_;
  dim3 gemm_grid(E_ / 64, M_ / 128), gemm_blk(256);
  dim3 attn_grid(S_ / 64, B_ * H_), attn_blk(128);

  auto conv = [&](const void* src, h16* dst, int n) {
    uaa_conv_f16_kernel<<<(n + 255) / 256, 256, 0, stream>>>((const float*)src, dst, n);
  };

  // Q projection
  conv(d_in[5], w16, nW);
  conv(d_in[0], x16, nX);
  uaa_gemm_kernel<<<gemm_grid, gemm_blk, 0, stream>>>(x16, w16, (const float*)d_in[6], qh, 0);
  // K projection
  conv(d_in[7], w16, nW);
  conv(d_in[1], x16, nX);
  uaa_gemm_kernel<<<gemm_grid, gemm_blk, 0, stream>>>(x16, w16, (const float*)d_in[8], kh, 0);
  // V projection (stored head-transposed for the PV B operand)
  conv(d_in[9], w16, nW);
  conv(d_in[2], x16, nX);
  uaa_gemm_kernel<<<gemm_grid, gemm_blk, 0, stream>>>(x16, w16, (const float*)d_in[10], vT, 1);
  // attention: writes attn (d_out) + attended f16 (into x16)
  uaa_attn_kernel<<<attn_grid, attn_blk, 0, stream>>>(qh, kh, vT, mask, x16, attn);
  // output projection
  conv(d_in[15], w16, nW);
  uaa_gemm_kernel<<<gemm_grid, gemm_blk, 0, stream>>>(x16, w16, (const float*)d_in[16], pre, 2);
  // gated residual
  uaa_gate_kernel<<<M_, 256, 0, stream>>>(pre, (const float*)d_in[17],
                                          (const float*)d_in[18], query, out);
}
